// MHSelfAttention_12214886990193
// MI455X (gfx1250) — matrix operand order repaired and checked by reading, not compiled, not run on hardware
//
#include <hip/hip_runtime.h>

// ---------------------------------------------------------------------------
// Types matching the gfx1250 WMMA builtin signatures
// ---------------------------------------------------------------------------
typedef __attribute__((ext_vector_type(16))) __bf16 bf16x16;
typedef __attribute__((ext_vector_type(8)))  float  f32x8;

union ABu    { bf16x16 v; uint4 u[2]; };           // 16 bf16 = 32 bytes
union U16x32 { uint4 u[4]; unsigned short s[32]; };

// float -> bf16 round-to-nearest-even
__device__ __forceinline__ unsigned short f2bf(float f) {
  unsigned int u = __float_as_uint(f);
  u = (u + 0x7FFFu + ((u >> 16) & 1u)) >> 16;
  return (unsigned short)u;
}

// ---------------------------------------------------------------------------
// Problem constants
// ---------------------------------------------------------------------------
#define BATCH 2
#define SEQ   2048
#define DIM   1024
#define HEADS 16
#define DH    64
#define ROWS  (BATCH * SEQ)       // 4096
#define QKVC  (3 * DIM)           // 3072

// ---------------------------------------------------------------------------
// Tensor Data Mover: async 8KB contiguous global->LDS copy (gfx1250 TDM).
// Descriptor per CDNA5 ISA. Guarded: falls back to the
// register-staged copy when the builtin is unavailable (incl. host pass).
// ---------------------------------------------------------------------------
#if defined(__has_builtin)
#if __has_builtin(__builtin_amdgcn_tensor_load_to_lds)
#define HAVE_TDM 1
#endif
#endif

#ifdef HAVE_TDM
typedef __attribute__((ext_vector_type(4))) unsigned int u32x4;
typedef __attribute__((ext_vector_type(8))) int          i32x8;
typedef __attribute__((ext_vector_type(4))) int          i32x4;

__device__ __forceinline__ void tdm_load_8kb(const void* gsrc, void* lds_dst) {
  unsigned long long ga = (unsigned long long)(uintptr_t)gsrc;
  unsigned int lo = (unsigned int)(uintptr_t)lds_dst;   // LDS aperture: addr[31:0]
  // D# group 0: count=1 (valid) | lds_addr | global_addr[56:0] | type=2
  u32x4 g0 = { 1u, lo, (unsigned int)ga,
               (unsigned int)((ga >> 32) & 0x1FFFFFFu) | (2u << 30) };
  const unsigned int td0   = 1u << 24;   // huge tensor_dim0 (8B units): no OOB clip
  const unsigned int tile0 = 1024u;      // 1024 x 8B = 8 KB tile, 1-D
  i32x8 g1 = { (int)(3u << 16),                              // data_size = 8B
               (int)((td0 & 0xFFFFu) << 16),                 // tensor_dim0[15:0]
               (int)(((td0 >> 16) & 0xFFFFu) | (1u << 16)),  // td0[31:16] | td1[15:0]=1
               (int)(tile0 << 16),                           // td1[31:16]=0 | tile_dim0
               (int)1,                                       // tile_dim1=1, tile_dim2=0
               (int)td0,                                     // tensor_dim0_stride lo32
               0, 0 };
  i32x4 gz = {0, 0, 0, 0};
#if __clang_major__ >= 23
  i32x8 gz8 = {0, 0, 0, 0, 0, 0, 0, 0};
  __builtin_amdgcn_tensor_load_to_lds(g0, g1, gz, gz, gz8, 0);
#else
  __builtin_amdgcn_tensor_load_to_lds(g0, g1, gz, gz, 0);
#endif
}
#endif

// ---------------------------------------------------------------------------
// Conversion kernels
// ---------------------------------------------------------------------------
__global__ void f32_to_bf16_kernel(const float* __restrict__ in,
                                   unsigned short* __restrict__ out, int n) {
  for (int i = blockIdx.x * blockDim.x + threadIdx.x; i < n;
       i += gridDim.x * blockDim.x)
    out[i] = f2bf(in[i]);
}

// in[rows][cols] (f32, row-major) -> out[cols][rows] (bf16)
__global__ void transpose_f32_to_bf16_kernel(const float* __restrict__ in,
                                             unsigned short* __restrict__ out,
                                             int rows, int cols) {
  int total = rows * cols;
  for (int i = blockIdx.x * blockDim.x + threadIdx.x; i < total;
       i += gridDim.x * blockDim.x) {
    int r = i / cols, c = i - r * cols;
    out[c * rows + r] = f2bf(in[i]);
  }
}

// ---------------------------------------------------------------------------
// Shared GEMM core: C(64x64 tile) = A[rows x DIM] * W (via Wt col-major).
// 128 threads = 4 waves x 16 rows.  Ping-pong LDS B tile (2 x 64x32 bf16),
// ONE barrier per K step: issue next-tile global loads first, compute from
// current buffer, publish staged regs into the other buffer, barrier.
// ---------------------------------------------------------------------------
__device__ __forceinline__ void gemm_tiles(
    const unsigned short* __restrict__ arow,  // A row for this lane (row = lo)
    const unsigned short* __restrict__ wp,    // Wt + (colbase+col)*DIM + off
    unsigned short* sB,                       // [2][64*32]
    int col, int off, int lo, int hi8, int hi16, f32x8 acc[4]) {
  // preload K-tile 0
  uint4 b0 = *reinterpret_cast<const uint4*>(wp);
  uint4 b1 = *reinterpret_cast<const uint4*>(wp + 8);
  uint4 a0 = *reinterpret_cast<const uint4*>(arow + hi8);
  uint4 a1 = *reinterpret_cast<const uint4*>(arow + 16 + hi8);
  {
    uint4* dst = reinterpret_cast<uint4*>(&sB[col * 32 + off]);
    dst[0] = b0;
    dst[1] = b1;
  }
  __syncthreads();

  const int NK = DIM / 32;
  for (int i = 0; i < NK; ++i) {
    uint4 nb0, nb1, na0, na1;
    const bool more = (i + 1) < NK;
    if (more) {   // issue next-tile global loads early (hide latency)
      const unsigned short* wn = wp + (i + 1) * 32;
      nb0 = *reinterpret_cast<const uint4*>(wn);
      nb1 = *reinterpret_cast<const uint4*>(wn + 8);
      const unsigned short* an = arow + (i + 1) * 32;
      na0 = *reinterpret_cast<const uint4*>(an + hi8);
      na1 = *reinterpret_cast<const uint4*>(an + 16 + hi8);
    }
    ABu a;
    a.u[0] = a0;
    a.u[1] = a1;
    const unsigned short* cur = sB + (i & 1) * (64 * 32);
    ABu b[4];
#pragma unroll
    for (int nt = 0; nt < 4; ++nt) {   // clause all LDS loads...
      const unsigned short* bp = cur + (nt * 16 + lo) * 32 + hi8;
      b[nt].u[0] = *reinterpret_cast<const uint4*>(bp);
      b[nt].u[1] = *reinterpret_cast<const uint4*>(bp + 16);
    }
#pragma unroll
    for (int nt = 0; nt < 4; ++nt)     // ...then WMMAs back-to-back
      acc[nt] = __builtin_amdgcn_wmma_f32_16x16x32_bf16(
          false, a.v, false, b[nt].v, (short)0, acc[nt], false, false);
    if (more) {  // publish next tile into the other buffer
      uint4* dst = reinterpret_cast<uint4*>(
          &sB[((i + 1) & 1) * (64 * 32) + col * 32 + off]);
      dst[0] = nb0;
      dst[1] = nb1;
      a0 = na0;
      a1 = na1;
    }
    __syncthreads();
  }
}

// ---------------------------------------------------------------------------
// QKV projection GEMM with fused scatter epilogue.  Reference factoring:
// col e = d*48 + sel*16 + h (sel 0=Q,1=K,2=V); SCALE folded into Q.
// ---------------------------------------------------------------------------
__global__ __launch_bounds__(128) void qkv_gemm_kernel(
    const unsigned short* __restrict__ xb,
    const unsigned short* __restrict__ wt,
    unsigned short* __restrict__ Q,
    unsigned short* __restrict__ K,
    unsigned short* __restrict__ V) {
  __shared__ unsigned short sB[2 * 64 * 32];

  const int tid  = threadIdx.x;
  const int lane = tid & 31;
  const int wave = tid >> 5;
  const int lo   = lane & 15;
  const int hi8  = (lane >> 4) << 3;
  const int hi16 = (lane >> 4) << 4;
  const int rowbase = blockIdx.x * 64;
  const int colbase = blockIdx.y * 64;
  const int col = tid >> 1;
  const int off = (tid & 1) << 4;

  f32x8 acc[4] = {};
  gemm_tiles(xb + (size_t)(rowbase + wave * 16 + lo) * DIM,
             wt + (size_t)(colbase + col) * DIM + off, sB, col, off, lo, hi8,
             hi16, acc);

  unsigned short* qkv[3] = {Q, K, V};
#pragma unroll
  for (int nt = 0; nt < 4; ++nt) {
    int jb  = colbase + nt * 16;  // 16-aligned -> d, sel constant; h = lo
    int d   = jb / 48;
    int sel = (jb / 16) % 3;
    float scale = (sel == 0) ? 0.125f : 1.0f;  // SCALE = DH^-0.5
    unsigned short* dst = qkv[sel];
#pragma unroll
    for (int r = 0; r < 8; ++r) {
      int grow = rowbase + wave * 16 + r + hi8;
      int bI = grow >> 11, nI = grow & (SEQ - 1);
      dst[(((size_t)(bI * HEADS + lo) * SEQ + nI) << 6) + d] =
          f2bf(acc[nt][r] * scale);
    }
  }
}

// ---------------------------------------------------------------------------
// Causal flash attention.  Grid: (N/64 query blocks, B*H).  4 waves x 16
// query rows.  Double-buffered K/V tiles; next K tile via TDM (async DMA,
// TENSORcnt) when available, else register-staged; V staged+transposed so
// the P@V B-operand is two 16B LDS reads.
// ---------------------------------------------------------------------------
__global__ __launch_bounds__(128) void attn_kernel(
    const unsigned short* __restrict__ Qg,
    const unsigned short* __restrict__ Kg,
    const unsigned short* __restrict__ Vg,
    unsigned short* __restrict__ attnb) {
  __shared__ unsigned short sK[2][64 * 64];   // [key][dh]
  __shared__ unsigned short sVt[2][64 * 64];  // [dh][key]
  __shared__ unsigned short sP[4 * 16 * 64];  // per-wave P staging

  const int tid  = threadIdx.x;
  const int lane = tid & 31;
  const int wave = tid >> 5;
  const int lo   = lane & 15;
  const int hi8  = (lane >> 4) << 3;
  const int hi16 = (lane >> 4) << 4;
  const int qb = blockIdx.x;
  const int bh = blockIdx.y;
  const int kr = tid >> 1, half = tid & 1;

  const unsigned short* kbase = Kg + (size_t)bh * SEQ * DH;
  const unsigned short* vbase = Vg + (size_t)bh * SEQ * DH;

  // Q tile (16 rows x 64) into A-operand registers, 2 chunks of K=32
  const unsigned short* qrow =
      Qg + ((size_t)bh * SEQ + qb * 64 + wave * 16 + lo) * DH;
  ABu qa[2];
#pragma unroll
  for (int c = 0; c < 2; ++c) {
    qa[c].u[0] = *reinterpret_cast<const uint4*>(qrow + c * 32 + hi8);
    qa[c].u[1] = *reinterpret_cast<const uint4*>(qrow + c * 32 + 16 + hi8);
  }

  f32x8 O[4] = {};
  float m_i[8], l_i[8];
#pragma unroll
  for (int r = 0; r < 8; ++r) { m_i[r] = -1e30f; l_i[r] = 0.f; }

  // ---- preload tile 0 into buffer 0 ----
  {
#ifdef HAVE_TDM
    if (tid == 0) tdm_load_8kb(kbase, &sK[0][0]);
#else
    const uint4* src = reinterpret_cast<const uint4*>(kbase);
    uint4* dst = reinterpret_cast<uint4*>(&sK[0][0]);
#pragma unroll
    for (int i = 0; i < 4; ++i) dst[tid * 4 + i] = src[tid * 4 + i];
#endif
    const uint4* vs = reinterpret_cast<const uint4*>(
        vbase + (size_t)kr * DH + half * 32);
    U16x32 vtmp;
#pragma unroll
    for (int i = 0; i < 4; ++i) vtmp.u[i] = vs[i];
#pragma unroll
    for (int i = 0; i < 32; ++i) sVt[0][(half * 32 + i) * 64 + kr] = vtmp.s[i];
#ifdef HAVE_TDM
    __builtin_amdgcn_s_wait_tensorcnt((short)0);
#endif
  }
  __syncthreads();

  for (int kt = 0; kt <= qb; ++kt) {
    const int cur = kt & 1, nxt = cur ^ 1;
    const bool more = kt < qb;

    // ---- stage tile kt+1 (loads issued before compute to hide latency) ----
    U16x32 vtmp;
#ifndef HAVE_TDM
    uint4 ktmp[4];
#endif
    if (more) {
#ifdef HAVE_TDM
      if (tid == 0) tdm_load_8kb(kbase + (size_t)(kt + 1) * 64 * DH,
                                 &sK[nxt][0]);
#else
      const uint4* ks =
          reinterpret_cast<const uint4*>(kbase + (size_t)(kt + 1) * 64 * DH);
#pragma unroll
      for (int i = 0; i < 4; ++i) ktmp[i] = ks[tid * 4 + i];
#endif
      const uint4* vs = reinterpret_cast<const uint4*>(
          vbase + (size_t)((kt + 1) * 64 + kr) * DH + half * 32);
#pragma unroll
      for (int i = 0; i < 4; ++i) vtmp.u[i] = vs[i];
    }

    // ---- S = Q K^T : 4 tiles of 16x16, 2 WMMAs each (dh split 32+32) ----
    f32x8 s[4];
#pragma unroll
    for (int t = 0; t < 4; ++t) {
      f32x8 z = {};
#pragma unroll
      for (int c = 0; c < 2; ++c) {
        ABu b;
        const unsigned short* bp = &sK[cur][(t * 16 + lo) * 64 + c * 32 + hi8];
        b.u[0] = *reinterpret_cast<const uint4*>(bp);
        b.u[1] = *reinterpret_cast<const uint4*>(bp + 16);
        z = __builtin_amdgcn_wmma_f32_16x16x32_bf16(
            false, qa[c].v, false, b.v, (short)0, z, false, false);
      }
      s[t] = z;
    }

    if (kt == qb) {  // causal mask only touches the diagonal block
#pragma unroll
      for (int t = 0; t < 4; ++t) {
        int kj = t * 16 + lo;
#pragma unroll
        for (int r = 0; r < 8; ++r)
          if (kj > wave * 16 + r + hi8) s[t][r] = -1e30f;
      }
    }

    // ---- online softmax; row M spans the 16 lanes of one half-wave ----
#pragma unroll
    for (int r = 0; r < 8; ++r) {
      float x = fmaxf(fmaxf(s[0][r], s[1][r]), fmaxf(s[2][r], s[3][r]));
#pragma unroll
      for (int o = 1; o < 16; o <<= 1) x = fmaxf(x, __shfl_xor(x, o, 32));
      float mn   = fmaxf(m_i[r], x);
      float corr = __expf(m_i[r] - mn);
      m_i[r] = mn;
      float rs = 0.f;
#pragma unroll
      for (int t = 0; t < 4; ++t) {
        float p = __expf(s[t][r] - mn);
        s[t][r] = p;
        rs += p;
      }
#pragma unroll
      for (int o = 1; o < 16; o <<= 1) rs += __shfl_xor(rs, o, 32);
      l_i[r] = l_i[r] * corr + rs;
#pragma unroll
      for (int t = 0; t < 4; ++t) O[t][r] *= corr;
    }

    // ---- C/D layout -> A layout via per-wave LDS staging (bf16) ----
    unsigned short* pw = &sP[wave * 16 * 64];
#pragma unroll
    for (int r = 0; r < 8; ++r) {
      int m = r + hi8;
#pragma unroll
      for (int t = 0; t < 4; ++t) pw[m * 64 + t * 16 + lo] = f2bf(s[t][r]);
    }

    // ---- O += P @ V : 2 key chunks x 4 dh tiles ----
#pragma unroll
    for (int c = 0; c < 2; ++c) {
      ABu pa;
      const unsigned short* pp = pw + lo * 64 + c * 32 + hi8;
      pa.u[0] = *reinterpret_cast<const uint4*>(pp);
      pa.u[1] = *reinterpret_cast<const uint4*>(pp + 16);
#pragma unroll
      for (int nt = 0; nt < 4; ++nt) {
        ABu b;
        const unsigned short* vp =
            &sVt[cur][(nt * 16 + lo) * 64 + c * 32 + hi8];
        b.u[0] = *reinterpret_cast<const uint4*>(vp);
        b.u[1] = *reinterpret_cast<const uint4*>(vp + 16);
        O[nt] = __builtin_amdgcn_wmma_f32_16x16x32_bf16(
            false, pa.v, false, b.v, (short)0, O[nt], false, false);
      }
    }

    // ---- publish staged tile kt+1 into the other buffer ----
    if (more) {
#ifndef HAVE_TDM
      uint4* kd = reinterpret_cast<uint4*>(&sK[nxt][0]);
#pragma unroll
      for (int i = 0; i < 4; ++i) kd[tid * 4 + i] = ktmp[i];
#endif
#pragma unroll
      for (int i = 0; i < 32; ++i)
        sVt[nxt][(half * 32 + i) * 64 + kr] = vtmp.s[i];
    }
#ifdef HAVE_TDM
    if (more) __builtin_amdgcn_s_wait_tensorcnt((short)0);
#endif
    __syncthreads();
  }

  // ---- normalize, store to attnb [B*N][H*DH] ----
  int bI = bh >> 4, h = bh & 15;
#pragma unroll
  for (int r = 0; r < 8; ++r) {
    float rl = 1.f / l_i[r];
    size_t grow = (size_t)bI * SEQ + qb * 64 + wave * 16 + r + hi8;
#pragma unroll
    for (int nt = 0; nt < 4; ++nt)
      attnb[grow * DIM + h * DH + nt * 16 + lo] = f2bf(O[nt][r] * rl);
  }
}

// ---------------------------------------------------------------------------
// Output projection: attnb[4096x1024] bf16 * Wout (via Wt[1024x1024]) -> f32
// ---------------------------------------------------------------------------
__global__ __launch_bounds__(128) void out_gemm_kernel(
    const unsigned short* __restrict__ ab,
    const unsigned short* __restrict__ wt,
    float* __restrict__ out) {
  __shared__ unsigned short sB[2 * 64 * 32];

  const int tid  = threadIdx.x;
  const int lane = tid & 31;
  const int wave = tid >> 5;
  const int lo   = lane & 15;
  const int hi8  = (lane >> 4) << 3;
  const int hi16 = (lane >> 4) << 4;
  const int rowbase = blockIdx.x * 64;
  const int colbase = blockIdx.y * 64;
  const int col = tid >> 1;
  const int off = (tid & 1) << 4;

  f32x8 acc[4] = {};
  gemm_tiles(ab + (size_t)(rowbase + wave * 16 + lo) * DIM,
             wt + (size_t)(colbase + col) * DIM + off, sB, col, off, lo, hi8,
             hi16, acc);

#pragma unroll
  for (int nt = 0; nt < 4; ++nt)
#pragma unroll
    for (int r = 0; r < 8; ++r)
      out[(size_t)(rowbase + wave * 16 + r + hi8) * DIM + colbase + nt * 16 +
          lo] = acc[nt][r];
}

// ---------------------------------------------------------------------------
// Host launch
// ---------------------------------------------------------------------------
extern "C" void kernel_launch(void* const* d_in, const int* in_sizes, int n_in,
                              void* d_out, int out_size, void* d_ws,
                              size_t ws_size, hipStream_t stream) {
  (void)in_sizes; (void)n_in; (void)out_size; (void)ws_size;
  const float* x    = (const float*)d_in[0];
  const float* Wqkv = (const float*)d_in[1];
  const float* Wout = (const float*)d_in[2];
  float* out = (float*)d_out;

  unsigned short* ws    = (unsigned short*)d_ws;
  unsigned short* xb    = ws;                                   // 4096*1024
  unsigned short* wqkvt = xb    + (size_t)ROWS * DIM;           // 3072*1024
  unsigned short* woutt = wqkvt + (size_t)QKVC * DIM;           // 1024*1024
  unsigned short* Q     = woutt + (size_t)DIM * DIM;            // 32*2048*64
  unsigned short* K     = Q + (size_t)BATCH * HEADS * SEQ * DH;
  unsigned short* V     = K + (size_t)BATCH * HEADS * SEQ * DH;
  unsigned short* attnb = V + (size_t)BATCH * HEADS * SEQ * DH; // 4096*1024

  f32_to_bf16_kernel<<<2048, 256, 0, stream>>>(x, xb, ROWS * DIM);
  transpose_f32_to_bf16_kernel<<<2048, 256, 0, stream>>>(Wqkv, wqkvt, DIM,
                                                         QKVC);
  transpose_f32_to_bf16_kernel<<<1024, 256, 0, stream>>>(Wout, woutt, DIM,
                                                         DIM);

  qkv_gemm_kernel<<<dim3(ROWS / 64, QKVC / 64), 128, 0, stream>>>(xb, wqkvt, Q,
                                                                  K, V);
  attn_kernel<<<dim3(SEQ / 64, BATCH * HEADS), 128, 0, stream>>>(Q, K, V,
                                                                 attnb);
  out_gemm_kernel<<<dim3(ROWS / 64, DIM / 64), 128, 0, stream>>>(attnb, woutt,
                                                                 out);
}
